// QuantumFeedForward_39960375722236
// MI455X (gfx1250) — matrix operand order repaired and checked by reading, not compiled, not run on hardware
//
#include <hip/hip_runtime.h>

// CDNA5 / gfx1250 FFN: fp32 GEMM via bf16x3 split-precision WMMA
// (aH*bH + aH*bL + aL*bH), fp32 accumulate. Double-buffered LDS fragment
// staging, software-pipelined global loads, compile-time strides.

typedef __attribute__((ext_vector_type(16))) __bf16 v16bf;
typedef __attribute__((ext_vector_type(4)))  __bf16 v4bf;
typedef __attribute__((ext_vector_type(8)))  float  v8f;
typedef __attribute__((ext_vector_type(4)))  float  v4f;

#define BLOCK_M 128
#define BLOCK_N 128
#define KSTEP   32
#define M_TOT   8192  // B*S = 4*2048

__device__ __forceinline__ void split4(const v4f v, v4bf& hi, v4bf& lo) {
    #pragma unroll
    for (int q = 0; q < 4; ++q) {
        __bf16 h = (__bf16)v[q];
        hi[q] = h;
        lo[q] = (__bf16)(v[q] - (float)h);
    }
}

template <int K, int N, bool FUSE_QUANTUM>
__global__ __launch_bounds__(256) void ffn_gemm_bf16x3(
    const float* __restrict__ A,      // [M_TOT x K]
    const float* __restrict__ Bm,     // [K x N]
    const float* __restrict__ bias,   // [N]
    const float* __restrict__ theta,  // [N]
    const float* __restrict__ wre,    // [N]
    const float* __restrict__ wim,    // [N]
    float* __restrict__ C)            // [M_TOT x N]
{
    // Double-buffered fragment-layout LDS: [buf][tile][lane][j] (32B/lane/tile)
    __shared__ __align__(32) __bf16 AsH[2][BLOCK_M / 16][32][16];
    __shared__ __align__(32) __bf16 AsL[2][BLOCK_M / 16][32][16];
    __shared__ __align__(32) __bf16 BsH[2][BLOCK_N / 16][32][16];
    __shared__ __align__(32) __bf16 BsL[2][BLOCK_N / 16][32][16];

    const int tid  = threadIdx.x;
    const int lane = tid & 31;
    const int wv   = tid >> 5;            // 8 waves
    const int m0   = blockIdx.y * BLOCK_M;
    const int n0   = blockIdx.x * BLOCK_N;
    const int mw   = (wv & 3) * 32;       // wave tile: 32 (M) x 64 (N)
    const int nw   = (wv >> 2) * 64;

    // ---- Per-thread staging geometry (all loop-invariant) ----
    // A: thread owns rows r = (tid>>3) + 32e, cols c4..c4+3 with c4=(tid&7)*4.
    const int r0   = tid >> 3;
    const int c4   = (tid & 7) << 2;
    const int a_sl = (r0 & 15) + ((c4 & 8) ? 16 : 0);
    const int a_sj = (c4 & 7) + ((c4 & 16) ? 8 : 0);
    const float* Aptr = A + (size_t)(m0 + r0) * K + c4;
    // B: thread owns col n = tid&127, K rows kk = 4t + 8e + q, t = tid>>7.
    const int bn   = tid & 127;
    const int bt   = tid >> 7;
    const float* Bptr = Bm + (size_t)(4 * bt) * N + (n0 + bn);

    v8f acc[2][4] = {};                   // 2x4 tiles of 16x16 fp32

    v4f  areg[4];
    float breg[16];

    // ---- Prologue: load tile 0 ----
    #pragma unroll
    for (int e = 0; e < 4; ++e)
        areg[e] = *reinterpret_cast<const v4f*>(Aptr + (size_t)e * 32 * K);
    #pragma unroll
    for (int e = 0; e < 4; ++e)
        #pragma unroll
        for (int q = 0; q < 4; ++q)
            breg[e * 4 + q] = Bptr[(size_t)(8 * e + q) * N];

    constexpr int NSTEPS = K / KSTEP;
    #pragma unroll 1
    for (int it = 0; it < NSTEPS; ++it) {
        const int buf = it & 1;

        // ---- Convert + store staged tile into fragment layout ----
        #pragma unroll
        for (int e = 0; e < 4; ++e) {
            v4bf hi, lo;
            split4(areg[e], hi, lo);
            const int tile = (r0 >> 4) + 2 * e;
            *reinterpret_cast<v4bf*>(&AsH[buf][tile][a_sl][a_sj]) = hi;
            *reinterpret_cast<v4bf*>(&AsL[buf][tile][a_sl][a_sj]) = lo;
        }
        #pragma unroll
        for (int e = 0; e < 4; ++e) {
            v4f bv; bv[0]=breg[e*4+0]; bv[1]=breg[e*4+1]; bv[2]=breg[e*4+2]; bv[3]=breg[e*4+3];
            v4bf hi, lo;
            split4(bv, hi, lo);
            const int tile = bn >> 4;
            const int sl   = (bn & 15) + ((e & 1) ? 16 : 0); // kk bit3 -> lane group
            const int sj   = 4 * bt + (e & 2) * 4;           // (kk & 7) + (kk bit4 ? 8 : 0)
            *reinterpret_cast<v4bf*>(&BsH[buf][tile][sl][sj]) = hi;
            *reinterpret_cast<v4bf*>(&BsL[buf][tile][sl][sj]) = lo;
        }

        __syncthreads();

        // ---- Software pipeline: issue next tile's global loads ----
        if (it + 1 < NSTEPS) {               // uniform branch, EXEC stays full
            const int k0n = (it + 1) * KSTEP;
            #pragma unroll
            for (int e = 0; e < 4; ++e)
                areg[e] = *reinterpret_cast<const v4f*>(Aptr + k0n + (size_t)e * 32 * K);
            #pragma unroll
            for (int e = 0; e < 4; ++e)
                #pragma unroll
                for (int q = 0; q < 4; ++q)
                    breg[e * 4 + q] = Bptr[(size_t)(k0n + 8 * e + q) * N];
            if (it + 2 < NSTEPS)
                __builtin_prefetch(Aptr + (it + 2) * KSTEP, 0, 1);
        }

        // ---- Fragment loads (contiguous 32B per operand) + WMMA ----
        v16bf aH[2], aL[2], bH[4], bL[4];
        #pragma unroll
        for (int i = 0; i < 2; ++i) {
            const int t = (mw >> 4) + i;
            aH[i] = *reinterpret_cast<const v16bf*>(&AsH[buf][t][lane][0]);
            aL[i] = *reinterpret_cast<const v16bf*>(&AsL[buf][t][lane][0]);
        }
        #pragma unroll
        for (int j = 0; j < 4; ++j) {
            const int t = (nw >> 4) + j;
            bH[j] = *reinterpret_cast<const v16bf*>(&BsH[buf][t][lane][0]);
            bL[j] = *reinterpret_cast<const v16bf*>(&BsL[buf][t][lane][0]);
        }

        #pragma unroll
        for (int i = 0; i < 2; ++i) {
            #pragma unroll
            for (int j = 0; j < 4; ++j) {
                acc[i][j] = __builtin_amdgcn_wmma_f32_16x16x32_bf16(
                    false, aH[i], false, bH[j], (short)0, acc[i][j], false, false);
                acc[i][j] = __builtin_amdgcn_wmma_f32_16x16x32_bf16(
                    false, aH[i], false, bL[j], (short)0, acc[i][j], false, false);
                acc[i][j] = __builtin_amdgcn_wmma_f32_16x16x32_bf16(
                    false, aL[i], false, bH[j], (short)0, acc[i][j], false, false);
            }
        }
    }

    // ---- Epilogue: C/D layout: M = r + ((lane&16)?8:0), N = lane&15 ----
    #pragma unroll
    for (int i = 0; i < 2; ++i) {
        #pragma unroll
        for (int j = 0; j < 4; ++j) {
            const int n  = n0 + nw + j * 16 + (lane & 15);
            const int mb = m0 + mw + i * 16 + ((lane & 16) ? 8 : 0);
            float* Cp = C + (size_t)mb * N + n;
            const float bn_v = bias[n];
            float th = 0.f, qp = 0.f, wi_v = 0.f;
            if (FUSE_QUANTUM) {
                th = theta[n];
                const float wr_v = wre[n];
                wi_v = wim[n];
                qp = __cosf(th) * wr_v + __sinf(th) * wi_v;
            }
            #pragma unroll
            for (int r = 0; r < 8; ++r) {
                float h = acc[i][j][r] + bn_v;
                float o;
                if (FUSE_QUANTUM) {
                    const float imag = wi_v * __sinf(th + h * 0.1f);
                    const float q = h + qp * imag * 0.1f;
                    o = fmaxf(q, 0.0f);     // ReLU (dropout = identity in eval)
                } else {
                    o = h;
                }
                Cp[(size_t)r * N] = o;
            }
        }
    }
}

extern "C" void kernel_launch(void* const* d_in, const int* in_sizes, int n_in,
                              void* d_out, int out_size, void* d_ws, size_t ws_size,
                              hipStream_t stream) {
    const float* x     = (const float*)d_in[0];  // [8192 x 1024]
    const float* W1    = (const float*)d_in[1];  // [1024 x 4096]
    const float* b1    = (const float*)d_in[2];  // [4096]
    const float* theta = (const float*)d_in[3];  // [4096]
    const float* wre   = (const float*)d_in[4];  // [4096]
    const float* wim   = (const float*)d_in[5];  // [4096]
    const float* W2    = (const float*)d_in[6];  // [4096 x 1024]
    const float* b2    = (const float*)d_in[7];  // [1024]

    float* act = (float*)d_ws;                   // [8192 x 4096] fp32
    float* out = (float*)d_out;                  // [8192 x 1024]

    dim3 block(256);

    // GEMM1 + bias + quantum transform + ReLU -> act
    dim3 grid1(4096 / BLOCK_N, M_TOT / BLOCK_M); // 32 x 64
    ffn_gemm_bf16x3<1024, 4096, true><<<grid1, block, 0, stream>>>(
        x, W1, b1, theta, wre, wim, act);

    // GEMM2 + bias -> out
    dim3 grid2(1024 / BLOCK_N, M_TOT / BLOCK_M); // 8 x 64
    ffn_gemm_bf16x3<4096, 1024, false><<<grid2, block, 0, stream>>>(
        act, W2, b2, nullptr, nullptr, nullptr, out);

    (void)in_sizes; (void)n_in; (void)out_size; (void)ws_size;
}
